// SS2D_16011638079739
// MI455X (gfx1250) — matrix operand order repaired and checked by reading, not compiled, not run on hardware
//
#include <hip/hip_runtime.h>
#include <hip/hip_bf16.h>
#include <math.h>

// ---------------- problem constants ----------------
#define CB   4
#define CH   64
#define CW   64
#define CL   4096      // H*W
#define CDM  128
#define CDI  256
#define CN   16
#define CR   8
#define CK   4
#define CC   40        // R + 2N
#define SCHUNK 16      // scan steps staged per LDS chunk

typedef __attribute__((ext_vector_type(16))) __bf16 v16bf;
typedef __attribute__((ext_vector_type(8)))  float  v8f;

// ---------------- optional gfx1250 async LDS path ----------------
#if defined(__gfx1250__) && __has_builtin(__builtin_amdgcn_global_load_async_to_lds_b32)
#define HAVE_ASYNC_LDS 1
#define GPTR(p) ((__attribute__((address_space(1))) int*)((void*)(p)))
#define LPTR(p) ((__attribute__((address_space(3))) int*)((void*)(p)))
#if __has_builtin(__builtin_amdgcn_s_wait_asynccnt)
#define WAIT_ASYNC() __builtin_amdgcn_s_wait_asynccnt(0)
#else
#define WAIT_ASYNC() asm volatile("s_wait_asynccnt 0x0" ::: "memory")
#endif
#endif

// ---------------- helpers ----------------
// Pack two f32 into one dword of 2x bf16 (lo, hi).
__device__ __forceinline__ unsigned pk_bf16(float lo, float hi) {
    union { __bf16 h[2]; unsigned u; } p;
    p.h[0] = (__bf16)lo; p.h[1] = (__bf16)hi;
    return p.u;
}

// WMMA fragment-linear LDS layout: each fragment is 32 lanes x 8 dwords (16 bf16/lane).
// A-matrix 16x32 (ISA 7.12.2): lane = m + (kk&8 ? 16:0); elem e = (kk&7) + (kk&16 ? 8:0).
__device__ __forceinline__ int a_slot(int m, int kk) {   // kk even; dword slot in 256-dword frag
    const int lane = m + ((kk & 8) ? 16 : 0);
    const int e    = (kk & 7) + ((kk & 16) ? 8 : 0);
    return lane * 8 + (e >> 1);
}
// B-matrix 32x16: lane = n + (kk&8 ? 16:0); elem e = (kk&7) + (kk&16 ? 8:0).
__device__ __forceinline__ int b_slot(int kk, int n) {   // kk even
    const int lane = n + ((kk & 8) ? 16 : 0);
    const int e    = (kk & 7) + ((kk & 16) ? 8 : 0);
    return lane * 8 + (e >> 1);
}

__device__ __forceinline__ v16bf frag_ld(const unsigned* base) {
    return *(const v16bf*)(base + ((threadIdx.x & 31) << 3));   // 32B aligned vector load
}

// Cross-scan index map: xs[k][d][l] == xconv[d][map_pos(k,l)]
__device__ __forceinline__ int map_pos(int k, int l) {
    int ll = (k & 2) ? (CL - 1 - l) : l;
    if (k & 1) {
        const int w = ll >> 6;     // l indexes (w,h); H=W=64
        const int h = ll & 63;
        return h * CW + w;
    }
    return ll;
}

// ================= kernel 1: in_proj GEMM =================
// out[r][d] = sum_c x[r][c]*W[d][c]; M=16384, K=128, N=256.
// block = 128 rows x 32 cols; wave = 16 rows x 32 cols (2 WMMA tiles).
__global__ void k_inproj(const float* __restrict__ x,
                         const float* __restrict__ w,
                         float* __restrict__ out) {
    __shared__ __align__(32) unsigned As[8 * 256];   // 8 A-fragments (16x32 each)
    __shared__ __align__(32) unsigned Bs[2 * 256];   // 2 B-fragments (32x16 each)
    const int tid  = threadIdx.x;
    const int wave = tid >> 5;
    const int rowBase = blockIdx.x * 128;
    const int colBase = blockIdx.y * 32;

    v8f acc0 = {}; v8f acc1 = {};

    for (int kc = 0; kc < CDM; kc += 32) {
        for (int i = tid; i < 128 * 16; i += 256) {          // A: pairs along K
            const int r  = i >> 4;
            const int c2 = (i & 15) << 1;
            const float* src = x + (size_t)(rowBase + r) * CDM + kc + c2;
            As[(r >> 4) * 256 + a_slot(r & 15, c2)] = pk_bf16(src[0], src[1]);
        }
        for (int i = tid; i < 32 * 16; i += 256) {           // B: pairs along K
            const int n  = i >> 4;
            const int c2 = (i & 15) << 1;
            const float* src = w + (size_t)(colBase + n) * CDM + kc + c2;
            Bs[(n >> 4) * 256 + b_slot(c2, n & 15)] = pk_bf16(src[0], src[1]);
        }
        __syncthreads();
        const v16bf a  = frag_ld(As + wave * 256);
        const v16bf b0 = frag_ld(Bs);
        const v16bf b1 = frag_ld(Bs + 256);
        acc0 = __builtin_amdgcn_wmma_f32_16x16x32_bf16(false, a, false, b0, (short)0, acc0, false, false);
        acc1 = __builtin_amdgcn_wmma_f32_16x16x32_bf16(false, a, false, b1, (short)0, acc1, false, false);
        __syncthreads();
    }

    const int lane = tid & 31;
    const int mrow = rowBase + wave * 16 + ((lane >= 16) ? 8 : 0);
    const int col  = colBase + (lane & 15);
#pragma unroll
    for (int v = 0; v < 8; ++v) {
        out[(size_t)(mrow + v) * CDI + col]      = acc0[v];
        out[(size_t)(mrow + v) * CDI + col + 16] = acc1[v];
    }
}

// ================= kernel 2: depthwise 3x3 conv + bias + SiLU =================
__global__ void k_conv(const float* __restrict__ xc_pre,
                       const float* __restrict__ cw,
                       const float* __restrict__ cb,
                       float* __restrict__ xconv) {
    const int id   = blockIdx.x * 256 + threadIdx.x;
    const int d    = id & 255;
    const int rest = id >> 8;
    const int w    = rest & 63;
    const int h    = (rest >> 6) & 63;
    const int b    = rest >> 12;

    float s = cb[d];
#pragma unroll
    for (int ky = 0; ky < 3; ++ky) {
#pragma unroll
        for (int kx = 0; kx < 3; ++kx) {
            const int hh = h + ky - 1, ww = w + kx - 1;
            if (hh >= 0 && hh < CH && ww >= 0 && ww < CW) {
                s += xc_pre[((size_t)b * CL + hh * CW + ww) * CDI + d] * cw[d * 9 + ky * 3 + kx];
            }
        }
    }
    const float sig = 1.0f / (1.0f + __expf(-s));
    xconv[((size_t)b * CDI + d) * CL + h * CW + w] = s * sig;
}

// ================= kernel 3: x_dbl GEMM (per b,k) =================
// x_dbl[b,k,c,l] = sum_d W[k,c,d]*xs[b,k,d,l]; rows 40 padded to 48.
// block = one (b,k) x 128 L-columns; wave = 48 rows x 16 cols (3 WMMA tiles).
__global__ void k_xdbl(const float* __restrict__ xconv,
                       const float* __restrict__ xprojw,
                       float* __restrict__ xdbl) {
    __shared__ __align__(32) unsigned As[3 * 256];   // 3 A-fragments
    __shared__ __align__(32) unsigned Bs[8 * 256];   // 8 B-fragments
    const int tid  = threadIdx.x;
    const int wave = tid >> 5;
    const int lane = tid & 31;
    const int bk   = blockIdx.x >> 5;
    const int lt   = blockIdx.x & 31;
    const int b    = bk >> 2, k = bk & 3;
    const int lBase = lt * 128;

    v8f acc0 = {}, acc1 = {}, acc2 = {};

    for (int kc = 0; kc < CDI; kc += 32) {
        for (int i = tid; i < 48 * 16; i += 256) {           // A: pairs along K
            const int m  = i >> 4;
            const int c2 = (i & 15) << 1;
            unsigned v = 0u;
            if (m < CC) {
                const float* src = xprojw + ((size_t)k * CC + m) * CDI + kc + c2;
                v = pk_bf16(src[0], src[1]);
            }
            As[(m >> 4) * 256 + a_slot(m & 15, c2)] = v;
        }
        for (int i = tid; i < 16 * 128; i += 256) {          // B: pairs along K (=d, stride CL)
            const int cc  = i & 127;
            const int kk2 = (i >> 7) << 1;
            const float* src = xconv + ((size_t)b * CDI + kc + kk2) * CL + map_pos(k, lBase + cc);
            Bs[(cc >> 4) * 256 + b_slot(kk2, cc & 15)] = pk_bf16(src[0], src[CL]);
        }
        __syncthreads();
        const v16bf bf = frag_ld(Bs + wave * 256);
        const v16bf a0 = frag_ld(As);
        const v16bf a1 = frag_ld(As + 256);
        const v16bf a2 = frag_ld(As + 512);
        acc0 = __builtin_amdgcn_wmma_f32_16x16x32_bf16(false, a0, false, bf, (short)0, acc0, false, false);
        acc1 = __builtin_amdgcn_wmma_f32_16x16x32_bf16(false, a1, false, bf, (short)0, acc1, false, false);
        acc2 = __builtin_amdgcn_wmma_f32_16x16x32_bf16(false, a2, false, bf, (short)0, acc2, false, false);
        __syncthreads();
    }

    const int col  = lBase + wave * 16 + (lane & 15);
    const int mofs = (lane >= 16) ? 8 : 0;
#pragma unroll
    for (int v = 0; v < 8; ++v) {
        const int m0 = 0  + v + mofs;
        const int m1 = 16 + v + mofs;
        const int m2 = 32 + v + mofs;
        xdbl[((size_t)bk * CC + m0) * CL + col] = acc0[v];
        xdbl[((size_t)bk * CC + m1) * CL + col] = acc1[v];
        if (m2 < CC) xdbl[((size_t)bk * CC + m2) * CL + col] = acc2[v];
    }
}

// ================= kernel 4: selective scan =================
// Block per (b,k); thread = channel d; 16 f32 states in VGPRs.
// Broadcast dts/Bs/Cs staged per SCHUNK steps in LDS ([step][48] layout -> b128 reads);
// staged via GLOBAL_LOAD_ASYNC_TO_LDS when available (ASYNCcnt path).
__global__ void k_scan(const float* __restrict__ xdbl,
                       const float* __restrict__ xconv,
                       const float* __restrict__ dtw_all,
                       const float* __restrict__ dtb_all,
                       const float* __restrict__ Alogs,
                       const float* __restrict__ Ds,
                       float* __restrict__ ys) {
    const int bk = blockIdx.x;
    const int b  = bk >> 2, k = bk & 3;
    const int d  = threadIdx.x;
    const int kd = k * CDI + d;

    float A[CN], h[CN], dtw[CR];
#pragma unroll
    for (int n = 0; n < CN; ++n) { A[n] = -__expf(Alogs[(size_t)kd * CN + n]); h[n] = 0.0f; }
#pragma unroll
    for (int r = 0; r < CR; ++r) dtw[r] = dtw_all[(size_t)kd * CR + r];
    const float dtb = dtb_all[kd];
    const float Dd  = Ds[kd];

    const float* xd    = xdbl + (size_t)bk * CC * CL;            // (40, L)
    const float* uchan = xconv + ((size_t)b * CDI + d) * CL;     // (L,)
    float* yrow        = ys + (size_t)bk * CL * CDI;             // (L, DI)

    __shared__ __align__(16) float sh[SCHUNK * 48];              // [step][48], rows 0..39 used

    for (int l0 = 0; l0 < CL; l0 += SCHUNK) {
        // ---- stage chunk (transposed) ----
#ifdef HAVE_ASYNC_LDS
        for (int i = threadIdx.x; i < CC * SCHUNK; i += 256) {
            const int row = i >> 4;        // 0..39
            const int lc  = i & (SCHUNK - 1);
            __builtin_amdgcn_global_load_async_to_lds_b32(
                GPTR(xd + (size_t)row * CL + l0 + lc),
                LPTR(sh + lc * 48 + row), 0, 0);
        }
        WAIT_ASYNC();
#else
        for (int i = threadIdx.x; i < CC * SCHUNK; i += 256) {
            const int row = i >> 4;
            const int lc  = i & (SCHUNK - 1);
            sh[lc * 48 + row] = xd[(size_t)row * CL + l0 + lc];
        }
#endif
        // ---- per-thread u values for this chunk (vectorized when contiguous) ----
        float ureg[SCHUNK];
        if ((k & 1) == 0) {
            const int p0 = (k & 2) ? (CL - SCHUNK - l0) : l0;
            float tmp[SCHUNK];
#pragma unroll
            for (int j = 0; j < SCHUNK / 4; ++j) {
                const float4 v = *(const float4*)(uchan + p0 + 4 * j);
                tmp[4*j] = v.x; tmp[4*j+1] = v.y; tmp[4*j+2] = v.z; tmp[4*j+3] = v.w;
            }
#pragma unroll
            for (int lc = 0; lc < SCHUNK; ++lc)
                ureg[lc] = (k & 2) ? tmp[SCHUNK - 1 - lc] : tmp[lc];
        } else {
#pragma unroll
            for (int lc = 0; lc < SCHUNK; ++lc)
                ureg[lc] = uchan[map_pos(k, l0 + lc)];
        }
        if (l0 + SCHUNK < CL) __builtin_prefetch(xd + l0 + SCHUNK, 0, 1);
        __syncthreads();

        // ---- SCHUNK sequential steps ----
#pragma unroll
        for (int lc = 0; lc < SCHUNK; ++lc) {
            const float* rp = sh + lc * 48;
            const float4 q0  = *(const float4*)(rp + 0);
            const float4 q1  = *(const float4*)(rp + 4);
            const float4 qb0 = *(const float4*)(rp + 8);
            const float4 qb1 = *(const float4*)(rp + 12);
            const float4 qb2 = *(const float4*)(rp + 16);
            const float4 qb3 = *(const float4*)(rp + 20);
            const float4 qc0 = *(const float4*)(rp + 24);
            const float4 qc1 = *(const float4*)(rp + 28);
            const float4 qc2 = *(const float4*)(rp + 32);
            const float4 qc3 = *(const float4*)(rp + 36);

            float dtraw = dtb
                + dtw[0]*q0.x + dtw[1]*q0.y + dtw[2]*q0.z + dtw[3]*q0.w
                + dtw[4]*q1.x + dtw[5]*q1.y + dtw[6]*q1.z + dtw[7]*q1.w;
            const float delta = (dtraw > 20.0f) ? dtraw : log1pf(__expf(dtraw));
            const float u  = ureg[lc];
            const float du = delta * u;

            const float Bv[CN] = {qb0.x,qb0.y,qb0.z,qb0.w, qb1.x,qb1.y,qb1.z,qb1.w,
                                  qb2.x,qb2.y,qb2.z,qb2.w, qb3.x,qb3.y,qb3.z,qb3.w};
            const float Cv[CN] = {qc0.x,qc0.y,qc0.z,qc0.w, qc1.x,qc1.y,qc1.z,qc1.w,
                                  qc2.x,qc2.y,qc2.z,qc2.w, qc3.x,qc3.y,qc3.z,qc3.w};
            float y = 0.0f;
#pragma unroll
            for (int n = 0; n < CN; ++n) {
                h[n] = __expf(delta * A[n]) * h[n] + du * Bv[n];
                y += h[n] * Cv[n];
            }
            yrow[(size_t)(l0 + lc) * CDI + d] = y + Dd * u;     // coalesced in d
        }
        __syncthreads();
    }
}

// ================= kernel 5: 4-direction merge + LayerNorm =================
__global__ void k_merge_ln(const float* __restrict__ ys,
                           const float* __restrict__ gamma,
                           const float* __restrict__ beta,
                           float* __restrict__ out) {
    const int bl = blockIdx.x;
    const int b  = bl >> 12;
    const int l  = bl & (CL - 1);
    const int hh = l >> 6, ww = l & 63;
    const int lt = ww * CH + hh;
    const int d  = threadIdx.x;

    const size_t base = (size_t)b * CK * CL * CDI;
    const float y = ys[base + ((size_t)0 * CL + l)            * CDI + d]
                  + ys[base + ((size_t)2 * CL + (CL - 1 - l))  * CDI + d]
                  + ys[base + ((size_t)1 * CL + lt)            * CDI + d]
                  + ys[base + ((size_t)3 * CL + (CL - 1 - lt)) * CDI + d];

    __shared__ float red[CDI];
    __shared__ float red2[CDI];
    red[d] = y; red2[d] = y * y;
    __syncthreads();
#pragma unroll
    for (int s = 128; s > 0; s >>= 1) {
        if (d < s) { red[d] += red[d + s]; red2[d] += red2[d + s]; }
        __syncthreads();
    }
    const float mu  = red[0] * (1.0f / CDI);
    const float var = red2[0] * (1.0f / CDI) - mu * mu;
    out[(size_t)bl * CDI + d] = (y - mu) * rsqrtf(var + 1e-5f) * gamma[d] + beta[d];
}

// ================= host launcher =================
extern "C" void kernel_launch(void* const* d_in, const int* in_sizes, int n_in,
                              void* d_out, int out_size, void* d_ws, size_t ws_size,
                              hipStream_t stream) {
    const float* x              = (const float*)d_in[0];
    const float* in_proj_w      = (const float*)d_in[1];
    const float* conv_w         = (const float*)d_in[2];
    const float* conv_b         = (const float*)d_in[3];
    const float* x_proj_weight  = (const float*)d_in[4];
    const float* dt_projs_w     = (const float*)d_in[5];
    const float* dt_projs_bias  = (const float*)d_in[6];
    const float* A_logs         = (const float*)d_in[7];
    const float* Ds             = (const float*)d_in[8];
    const float* ln_gamma       = (const float*)d_in[9];
    const float* ln_beta        = (const float*)d_in[10];
    float* out = (float*)d_out;

    float* ws      = (float*)d_ws;
    float* xc_pre  = ws;                                        // B*L*DI
    float* xconv   = xc_pre + (size_t)CB * CL * CDI;            // B*DI*L
    float* xdbl    = xconv  + (size_t)CB * CDI * CL;            // B*K*40*L
    float* ys      = xdbl   + (size_t)CB * CK * CC * CL;        // B*K*L*DI

    dim3 g1((CB * CL) / 128, CDI / 32);
    k_inproj<<<g1, 256, 0, stream>>>(x, in_proj_w, xc_pre);

    k_conv<<<(CB * CL * CDI) / 256, 256, 0, stream>>>(xc_pre, conv_w, conv_b, xconv);

    k_xdbl<<<CB * CK * 32, 256, 0, stream>>>(xconv, x_proj_weight, xdbl);

    k_scan<<<CB * CK, 256, 0, stream>>>(xdbl, xconv, dt_projs_w, dt_projs_bias,
                                        A_logs, Ds, ys);

    k_merge_ln<<<CB * CL, 256, 0, stream>>>(ys, ln_gamma, ln_beta, out);
}
